// FNO1d_52063593562979
// MI455X (gfx1250) — matrix operand order repaired and checked by reading, not compiled, not run on hardware
//
#include <hip/hip_runtime.h>
#include <math.h>

// ---------------- FNO1d constants ----------------
#define MODES 32
#define WIDTH 128
#define NLAYERS 4
#define BB 16
#define NN 8192
#define K2 (2*MODES)   // 64 basis rows (cos 0..31, sin 0..31)
#define KSPLIT 16      // split-K factor for the forward DFT GEMM

typedef __bf16 bf16;
typedef __bf16 v16bf __attribute__((ext_vector_type(16)));
typedef float  v8f   __attribute__((ext_vector_type(8)));

union AFrag { v16bf v; unsigned u[8]; };

// ---- WMMA fragment loaders (CDNA5 ISA 7.12.2 layouts, wave32) ----
// A (16-bit, 16x32): lane%16 = M row, lane/16 splits K; VGPR v holds packed
// bf16 pair at K = (v<4 ? 2v : 16+2(v-4)) + 8*(lane/16).
__device__ __forceinline__ v16bf load_a_frag(const bf16* As, int ldA, int mO, int lane) {
    int row  = mO + (lane & 15);
    int half = lane >> 4;
    AFrag f;
#pragma unroll
    for (int v = 0; v < 8; ++v) {
        int k = (v < 4 ? 2*v : 16 + 2*(v-4)) + 8*half;
        f.u[v] = *(const unsigned*)(As + row*ldA + k);
    }
    return f.v;
}
// B (16-bit, 32x16): lane%16 = N column, lane/16 splits K; element s at
// K = (s<8 ? s : 16+(s-8)) + 8*(lane/16) (same order as A).
__device__ __forceinline__ v16bf load_b_frag(const bf16* Bs, int ldB, int nO, int lane) {
    AFrag f;
    int col = nO + (lane & 15), half = lane >> 4;
#pragma unroll
    for (int s = 0; s < 16; ++s)
        f.v[s] = Bs[((s < 8 ? s : 8 + s) + 8*half)*ldB + col];
    return f.v;
}

__device__ __forceinline__ v8f wmma_bf16(v16bf a, v16bf b, v8f c) {
    // (neg_a, A, neg_b, B, c_mod, C, reuse_a, reuse_b)
    return __builtin_amdgcn_wmma_f32_16x16x32_bf16(false, a, false, b, (short)0, c, false, false);
}

__device__ __forceinline__ float gelu_exact(float x) {
    return 0.5f * x * (1.0f + erff(x * 0.70710678118654752f));
}

// ---------------- zero fill (split-K accumulator init) ----------------
__global__ __launch_bounds__(256) void k_zero(float* __restrict__ p) {
    p[blockIdx.x * 256 + threadIdx.x] = 0.f;
}

// ---------------- basis: rows 0..31 cos(2*pi*m*n/N), 32..63 sin ----------------
__global__ __launch_bounds__(256) void k_basis(bf16* __restrict__ basis) {
    int idx = blockIdx.x * 256 + threadIdx.x;          // 64*8192
    int n = idx & (NN - 1);
    int r = idx >> 13;
    int m = r & 31;
    int p = (2 * m * n) & (2 * NN - 1);                // exact mod-2N reduction
    float ang = (float)p * 3.834951969714103e-4f;      // pi/8192
    basis[idx] = (bf16)(r < 32 ? cosf(ang) : sinf(ang));
}

// ---------------- fc0: h[b][i][n] = x[b][n]*w[i] + b[i] ----------------
__global__ __launch_bounds__(256) void k_fc0(const float* __restrict__ x,
                                             const float* __restrict__ w,
                                             const float* __restrict__ bias,
                                             bf16* __restrict__ h) {
    size_t idx = (size_t)blockIdx.x * 256 + threadIdx.x;   // B*W*N
    int n = (int)(idx & (NN - 1));
    int i = (int)((idx >> 13) & (WIDTH - 1));
    int b = (int)(idx >> 20);
    h[idx] = (bf16)(x[((size_t)b << 13) + n] * w[i] + bias[i]);
}

// ---------------- forward truncated DFT (split-K) ----------------
// X[bi][r] += sum_{n in K-range} h[bi][n]*basis[r][n].
// GEMM M=2048, Ncols=64, K=8192 -> 16-way split-K, block tile 128x64,
// wave tile 32x32 (4 accumulators), 8 WMMAs per barrier.
__global__ __launch_bounds__(256) void k_fwd_dft(const bf16* __restrict__ h,
                                                 const bf16* __restrict__ basis,
                                                 float* __restrict__ X) {
    __shared__ bf16 As[128][66];
    __shared__ bf16 Bs[64][66];
    int tid = threadIdx.x, lane = tid & 31, wave = tid >> 5;
    int bi0 = blockIdx.x * 128;
    int kbase = blockIdx.y * (NN / KSPLIT);            // 512-wide K range
    int mO = (wave >> 1) * 32, nO = (wave & 1) * 32;
    v8f a00 = {}, a01 = {}, a10 = {}, a11 = {};
#pragma unroll 1
    for (int kk = kbase; kk < kbase + NN / KSPLIT; kk += 64) {
        {   // stage A: 128 rows x 64 K, 16 coalesced dwords/thread
            int row = tid >> 1, cb = (tid & 1) * 32;
            const unsigned* src = (const unsigned*)(h + (size_t)(bi0 + row) * NN + kk + cb);
            unsigned* dst = (unsigned*)&As[row][cb];
#pragma unroll
            for (int j = 0; j < 16; ++j) dst[j] = src[j];
            if (kk + 64 < kbase + NN / KSPLIT)   // next-slab prefetch
                __builtin_prefetch(h + (size_t)(bi0 + row) * NN + kk + 64 + cb, 0, 1);
        }
        {   // stage B transposed: Bs[k][r] = basis[r][kk+k]
            int r = tid >> 2, kb = (tid & 3) * 16;
            const bf16* src = basis + (size_t)r * NN + kk + kb;
#pragma unroll
            for (int j = 0; j < 16; ++j) Bs[kb + j][r] = src[j];
        }
        __syncthreads();
#pragma unroll
        for (int ku = 0; ku < 64; ku += 32) {
            v16bf b0 = load_b_frag(&Bs[ku][0], 66, nO, lane);
            v16bf b1 = load_b_frag(&Bs[ku][0], 66, nO + 16, lane);
            v16bf f0 = load_a_frag(&As[0][ku], 66, mO, lane);
            v16bf f1 = load_a_frag(&As[0][ku], 66, mO + 16, lane);
            a00 = wmma_bf16(f0, b0, a00);
            a01 = wmma_bf16(f0, b1, a01);
            a10 = wmma_bf16(f1, b0, a10);
            a11 = wmma_bf16(f1, b1, a11);
        }
        __syncthreads();
    }
    int c0 = nO + (lane & 15), half = lane >> 4;
#pragma unroll
    for (int v = 0; v < 8; ++v) {
        int r0 = bi0 + mO + v + 8 * half;
        atomicAdd(&X[(size_t)r0 * K2 + c0],            a00[v]);
        atomicAdd(&X[(size_t)r0 * K2 + c0 + 16],       a01[v]);
        atomicAdd(&X[(size_t)(r0 + 16) * K2 + c0],     a10[v]);
        atomicAdd(&X[(size_t)(r0 + 16) * K2 + c0 + 16], a11[v]);
    }
}

// ---------------- complex mode mixing + irfft coefficient scaling ----------------
// X holds C = sum h*cos (cols 0..31) and S = sum h*sin (cols 32..63); rfft X = C - iS.
// F = X*(wr+i wi): Fr = C*wr + S*wi ; Fi = C*wi - S*wr.
// irfft(32 modes): x[n] = (1/N)[Fr0 + 2*sum_{m>=1}(Frm cos - Fim sin)]
//  -> cosCoef[0]=Fr0/N, cosCoef[m]=2Frm/N, sinCoef[m]=-2Fim/N.
__global__ __launch_bounds__(256) void k_modemix(const float* __restrict__ X,
                                                 const float* __restrict__ wr,
                                                 const float* __restrict__ wi,
                                                 bf16* __restrict__ coef, int layer) {
    int t = blockIdx.x * 256 + threadIdx.x;     // B*W*MODES = 65536
    int m = t & 31, o = (t >> 5) & 127, b = t >> 12;
    const float* xb  = X + (size_t)b * WIDTH * K2;
    const float* wrp = wr + (size_t)layer * WIDTH * WIDTH * MODES + (size_t)o * MODES + m;
    const float* wip = wi + (size_t)layer * WIDTH * WIDTH * MODES + (size_t)o * MODES + m;
    float Fr = 0.f, Fi = 0.f;
#pragma unroll 4
    for (int i = 0; i < WIDTH; ++i) {
        float C = xb[i * K2 + m], S = xb[i * K2 + 32 + m];
        float a = wrp[(size_t)i * (WIDTH * MODES)];
        float c = wip[(size_t)i * (WIDTH * MODES)];
        Fr += C * a + S * c;
        Fi += C * c - S * a;
    }
    const float inv = 1.0f / (float)NN;
    float cosC = (m == 0) ? Fr * inv : 2.f * Fr * inv;
    float sinC = (m == 0) ? 0.f      : -2.f * Fi * inv;
    bf16* cp = coef + (size_t)(b * WIDTH + o) * K2;
    cp[m]      = (bf16)cosC;
    cp[32 + m] = (bf16)sinC;
}

// ---------------- fused irfft + skip conv + bias + gelu ----------------
// out[b][o][n] = act( sum_{r<64} coef[b][o][r]*basis[r][n]
//                   + sum_{i<128} skip_w[l][i][o]*h[b][i][n] + skip_b[l][o] )
// Concatenated K = 64 + 128 = 192 -> 3 K-slabs of 64. Block tile 64 o x 128 n,
// wave tile 32x32 (4 accumulators), 8 WMMAs per barrier.
__global__ __launch_bounds__(256) void k_layer(const bf16* __restrict__ coef,
                                               const bf16* __restrict__ basis,
                                               const bf16* __restrict__ hin,
                                               const float* __restrict__ skip_w,
                                               const float* __restrict__ skip_b,
                                               bf16* __restrict__ out,
                                               int layer, int apply_gelu, int transpose_out) {
    __shared__ bf16 As[64][66];
    __shared__ bf16 Bs[64][132];
    int tid = threadIdx.x, lane = tid & 31, wave = tid >> 5;
    int b = blockIdx.z, o0 = blockIdx.y * 64, n0 = blockIdx.x * 128;
    int mO = (wave >> 2) * 32, nO = (wave & 3) * 32;
    v8f a00 = {}, a01 = {}, a10 = {}, a11 = {};
#pragma unroll
    for (int ks = 0; ks < 3; ++ks) {
        int kk = ks * 64;
        if (ks == 0) {   // A <- coef[b][o][0..63], contiguous in k
            int oL = tid >> 2, cb = (tid & 3) * 16;
            const unsigned* src = (const unsigned*)(coef + (size_t)(b * WIDTH + o0 + oL) * K2 + cb);
            unsigned* dst = (unsigned*)&As[oL][cb];
#pragma unroll
            for (int j = 0; j < 8; ++j) dst[j] = src[j];
        } else {         // A <- transposed skip_w[l][ki][o] (coalesced along o)
            int kL = tid >> 2, ob = (tid & 3) * 16;
            int ki = kk - K2 + kL;
            const float* src = skip_w + ((size_t)layer * WIDTH + ki) * WIDTH + o0 + ob;
#pragma unroll
            for (int j = 0; j < 16; ++j) As[ob + j][kL] = (bf16)src[j];
        }
        {   // B: basis rows (slab 0) or h rows (slabs 1-2), contiguous in n
            int kL = tid >> 2, nb = (tid & 3) * 32;
            const bf16* src = (ks == 0)
                ? basis + (size_t)(kk + kL) * NN + n0 + nb
                : hin   + ((size_t)b * WIDTH + (kk - K2 + kL)) * NN + n0 + nb;
            const unsigned* s4 = (const unsigned*)src;
            unsigned* dq = (unsigned*)&Bs[kL][nb];
#pragma unroll
            for (int j = 0; j < 16; ++j) dq[j] = s4[j];
            if (ks < 2)   // prefetch next slab of the streaming operand
                __builtin_prefetch(hin + ((size_t)b * WIDTH + (kk - K2 + 64 + kL)) * NN + n0 + nb, 0, 1);
        }
        __syncthreads();
#pragma unroll
        for (int ku = 0; ku < 64; ku += 32) {
            v16bf b0 = load_b_frag(&Bs[ku][0], 132, nO, lane);
            v16bf b1 = load_b_frag(&Bs[ku][0], 132, nO + 16, lane);
            v16bf f0 = load_a_frag(&As[0][ku], 66, mO, lane);
            v16bf f1 = load_a_frag(&As[0][ku], 66, mO + 16, lane);
            a00 = wmma_bf16(f0, b0, a00);
            a01 = wmma_bf16(f0, b1, a01);
            a10 = wmma_bf16(f1, b0, a10);
            a11 = wmma_bf16(f1, b1, a11);
        }
        __syncthreads();
    }
    int half = lane >> 4;
#pragma unroll
    for (int v = 0; v < 8; ++v) {
#pragma unroll
        for (int mt = 0; mt < 2; ++mt) {
#pragma unroll
            for (int nt = 0; nt < 2; ++nt) {
                int o   = o0 + mO + 16 * mt + v + 8 * half;
                int col = n0 + nO + 16 * nt + (lane & 15);
                v8f* accp = mt ? (nt ? &a11 : &a10) : (nt ? &a01 : &a00);
                float val = (*accp)[v] + skip_b[layer * WIDTH + o];
                if (apply_gelu) val = gelu_exact(val);
                if (!transpose_out)
                    out[((size_t)b * WIDTH + o) * NN + col] = (bf16)val;   // [B,W,N]
                else
                    out[((size_t)b * NN + col) * WIDTH + o] = (bf16)val;   // [B,N,W] for fc1
            }
        }
    }
}

// ---------------- fc1 + gelu + fc2 fused ----------------
// 64 rows (b*N+n) x 128 cols per block; wave tile 32x32 (4 accumulators);
// K=128 in 2 slabs of 64 -> 8 WMMAs per barrier. Then per-row reduction of
// gelu(.)*fc2_w via LDS atomics.
__global__ __launch_bounds__(256) void k_fc12(const bf16* __restrict__ hT,
                                              const float* __restrict__ fc1_w,
                                              const float* __restrict__ fc1_b,
                                              const float* __restrict__ fc2_w,
                                              const float* __restrict__ fc2_b,
                                              float* __restrict__ outp) {
    __shared__ bf16 As[64][66];
    __shared__ bf16 Bs[64][132];
    __shared__ float rowAcc[64];
    int tid = threadIdx.x, lane = tid & 31, wave = tid >> 5;
    size_t bn0 = (size_t)blockIdx.x * 64;
    if (tid < 64) rowAcc[tid] = 0.f;
    int mO = (wave >> 2) * 32, nO = (wave & 3) * 32;
    v8f a00 = {}, a01 = {}, a10 = {}, a11 = {};
#pragma unroll
    for (int kk = 0; kk < WIDTH; kk += 64) {
        {   // A: 64 rows x 64 K from hT (contiguous rows)
            int row = tid >> 2, cb = (tid & 3) * 16;
            const unsigned* src = (const unsigned*)(hT + (bn0 + row) * WIDTH + kk + cb);
            unsigned* dst = (unsigned*)&As[row][cb];
#pragma unroll
            for (int j = 0; j < 8; ++j) dst[j] = src[j];
        }
        {   // B: fc1_w f32 -> bf16, 64 K x 128 cols
            int kL = tid >> 2, jb = (tid & 3) * 32;
            const float* src = fc1_w + (size_t)(kk + kL) * 128 + jb;
#pragma unroll
            for (int j = 0; j < 32; ++j) Bs[kL][jb + j] = (bf16)src[j];
        }
        __syncthreads();
#pragma unroll
        for (int ku = 0; ku < 64; ku += 32) {
            v16bf b0 = load_b_frag(&Bs[ku][0], 132, nO, lane);
            v16bf b1 = load_b_frag(&Bs[ku][0], 132, nO + 16, lane);
            v16bf f0 = load_a_frag(&As[0][ku], 66, mO, lane);
            v16bf f1 = load_a_frag(&As[0][ku], 66, mO + 16, lane);
            a00 = wmma_bf16(f0, b0, a00);
            a01 = wmma_bf16(f0, b1, a01);
            a10 = wmma_bf16(f1, b0, a10);
            a11 = wmma_bf16(f1, b1, a11);
        }
        __syncthreads();
    }
    int half = lane >> 4;
#pragma unroll
    for (int nt = 0; nt < 2; ++nt) {
        int j = nO + 16 * nt + (lane & 15);
        float fb = fc1_b[j], w2 = fc2_w[j];
#pragma unroll
        for (int v = 0; v < 8; ++v) {
            int r0 = mO + v + 8 * half;
            v8f* p0 = nt ? &a01 : &a00;
            v8f* p1 = nt ? &a11 : &a10;
            atomicAdd(&rowAcc[r0],      gelu_exact((*p0)[v] + fb) * w2);
            atomicAdd(&rowAcc[r0 + 16], gelu_exact((*p1)[v] + fb) * w2);
        }
    }
    __syncthreads();
    if (tid < 64) outp[bn0 + tid] = rowAcc[tid] + fc2_b[0];
}

// ---------------- host launch ----------------
extern "C" void kernel_launch(void* const* d_in, const int* in_sizes, int n_in,
                              void* d_out, int out_size, void* d_ws, size_t ws_size,
                              hipStream_t stream) {
    (void)in_sizes; (void)n_in; (void)out_size; (void)ws_size;
    const float* x       = (const float*)d_in[0];
    const float* fc0_w   = (const float*)d_in[1];
    const float* fc0_b   = (const float*)d_in[2];
    const float* spec_wr = (const float*)d_in[3];
    const float* spec_wi = (const float*)d_in[4];
    const float* skip_w  = (const float*)d_in[5];
    const float* skip_b  = (const float*)d_in[6];
    const float* fc1_w   = (const float*)d_in[7];
    const float* fc1_b   = (const float*)d_in[8];
    const float* fc2_w   = (const float*)d_in[9];
    const float* fc2_b   = (const float*)d_in[10];

    char* ws = (char*)d_ws;
    const size_t H_BYTES = (size_t)BB * WIDTH * NN * sizeof(bf16);   // 33554432
    bf16* h0    = (bf16*)(ws);
    bf16* h1    = (bf16*)(ws + H_BYTES);
    bf16* basis = (bf16*)(ws + 2 * H_BYTES);                          // 64*8192 bf16
    float* X    = (float*)(ws + 2 * H_BYTES + (size_t)K2 * NN * sizeof(bf16));
    bf16* coef  = (bf16*)((char*)X + (size_t)BB * WIDTH * K2 * sizeof(float));

    k_basis<<<(K2 * NN) / 256, 256, 0, stream>>>(basis);
    k_fc0<<<(BB * WIDTH * NN) / 256, 256, 0, stream>>>(x, fc0_w, fc0_b, h0);

    bf16* hin = h0;
    bf16* hout = h1;
    for (int l = 0; l < NLAYERS; ++l) {
        int last = (l == NLAYERS - 1);
        k_zero<<<(BB * WIDTH * K2) / 256, 256, 0, stream>>>(X);      // split-K init
        k_fwd_dft<<<dim3((BB * WIDTH) / 128, KSPLIT), 256, 0, stream>>>(hin, basis, X);
        k_modemix<<<(BB * WIDTH * MODES) / 256, 256, 0, stream>>>(X, spec_wr, spec_wi, coef, l);
        k_layer<<<dim3(NN / 128, WIDTH / 64, BB), 256, 0, stream>>>(
            coef, basis, hin, skip_w, skip_b, hout, l, !last, last);
        bf16* t = hin; hin = hout; hout = t;
    }
    // hin now holds layer-3 output in [B,N,W] layout
    k_fc12<<<(BB * NN) / 64, 256, 0, stream>>>(hin, fc1_w, fc1_b, fc2_w, fc2_b, (float*)d_out);
}
